// Kernel_62285615727205
// MI455X (gfx1250) — matrix operand order repaired and checked by reading, not compiled, not run on hardware
//
#include <hip/hip_runtime.h>
#include <hip/hip_bf16.h>

// ---------------------------------------------------------------------------
// Causal conv y[b,c,t] = sum_{s<=t} khat[c,s] * x[b,c,t-s]  (== reference
// zero-padded-FFT conv truncated to L), khat = squash(smooth(kern)).
// Block-Toeplitz matmul on v_wmma_f32_16x16x32_bf16:
//   A_delta[m,d'] = khat[16*delta + m - d']   (16x32, built from reversed LDS)
//   B_jp[d',n]    = x_bf16[batch_n, 32*jp+d'] (32x16, 8 real cols)
//   Y_i += A_{i-2jp} * B_jp  over all jp with i-2jp >= 0
// Each wave owns 8 output tiles; main loops are unguarded (delta <= iG) with
// B-fragment register rotation (1 new 32B load / iteration); k̂ row staged to
// LDS via the Tensor Data Mover.
// ---------------------------------------------------------------------------

typedef __attribute__((ext_vector_type(16))) __bf16 v16bf;
typedef __attribute__((ext_vector_type(8)))  float  v8f;
typedef __attribute__((ext_vector_type(4)))  unsigned int v4u;
typedef __attribute__((ext_vector_type(4)))  int    v4i;
typedef __attribute__((ext_vector_type(8)))  int    v8i;

#define C_CH   256
#define LEN    8192
#define BATCH  8
#define SM_P   3
#define KLAM   0.003f
#define KPAD   32
#define KTOT   (LEN + 2 * KPAD)

#if defined(__gfx1250__) && __has_builtin(__builtin_amdgcn_tensor_load_to_lds)
#define HAVE_TDM 1
#endif

// ---- kernel preprocessing: reflect-pad moving average (w=7) + soft-threshold,
// ---- written out REVERSED as bf16: kbr[c][j] = khat[c][LEN-1-j] -------------
__global__ void prep_kern(const float* __restrict__ kern, __bf16* __restrict__ kbr) {
    int idx = blockIdx.x * blockDim.x + threadIdx.x;
    if (idx >= C_CH * LEN) return;
    int c = idx / LEN, t = idx % LEN;
    const float* row = kern + (size_t)c * LEN;
    float s = 0.f;
#pragma unroll
    for (int d = -SM_P; d <= SM_P; ++d) {
        int u = t + d;
        if (u < 0)    u = -u;               // jnp reflect (no edge repeat)
        if (u >= LEN) u = 2 * LEN - 2 - u;
        s += row[u];
    }
    s *= (1.0f / (2 * SM_P + 1));
    float a = fabsf(s) - KLAM;
    a = a > 0.f ? a : 0.f;
    float r = (s > 0.f) ? a : ((s < 0.f) ? -a : 0.f);
    kbr[(size_t)c * LEN + (LEN - 1 - t)] = (__bf16)r;
}

// ---- x fp32 -> bf16 (fits whole tensor in the 192MB L2 for B re-reads) -----
__global__ void cvt_x(const float* __restrict__ x, __bf16* __restrict__ xb, int n4) {
    int i = (blockIdx.x * blockDim.x + threadIdx.x) * 4;
    if (i >= n4) return;
    float4 v = *(const float4*)(x + i);
    xb[i + 0] = (__bf16)v.x;
    xb[i + 1] = (__bf16)v.y;
    xb[i + 2] = (__bf16)v.z;
    xb[i + 3] = (__bf16)v.w;
}

// ---- main conv: grid (C, 8), block 256 (8 waves). Wave w owns 8 output
// ---- tiles i = iG..iG+7. ---------------------------------------------------
__launch_bounds__(256)
__global__ void conv_kern(const __bf16* __restrict__ xb,
                          const __bf16* __restrict__ kbr,
                          float* __restrict__ out) {
    // ksr[q] = khat[LEN + KPAD - 1 - q], zero outside [0,LEN): reversed+padded
    __shared__ __bf16 ksr[KTOT];

    const int c   = blockIdx.x;
    const int tid = threadIdx.x;

#if HAVE_TDM
    if (tid < KPAD) {                       // zero both pads
        ksr[tid]              = (__bf16)0.0f;
        ksr[LEN + KPAD + tid] = (__bf16)0.0f;
    }
    if (tid < 32) {                         // wave 0 issues one 16KB TDM DMA
        unsigned int lds_addr = (unsigned int)(size_t)(void*)&ksr[KPAD];
        unsigned long long ga = (unsigned long long)(size_t)(kbr + (size_t)c * LEN);
        // D# group0: count=1 | lds_addr | global_addr[56:0] | type=2
        v4u g0 = { 1u, lds_addr, (unsigned int)ga,
                   (unsigned int)((ga >> 32) & 0x1ffffffu) | 0x80000000u };
        // D# group1: data_size=4B; tensor_dim0=4096; tensor_dim1=1;
        //            tile_dim0=4096; tile_dim1=1; dim0_stride=4096
        v8i g1 = { 0x20000, 0x10000000, 0x10000, 0x10000000, 1, 4096, 0, 0 };
        v4i z4 = { 0, 0, 0, 0 };
#if __has_include(<hip/amd_detail/amd_gfx1250_TDM.h>)
        v8i z8 = { 0, 0, 0, 0, 0, 0, 0, 0 };
        __builtin_amdgcn_tensor_load_to_lds(g0, g1, z4, z4, z8, 0);
#else
        __builtin_amdgcn_tensor_load_to_lds(g0, g1, z4, z4, 0);
#endif
        __builtin_amdgcn_s_wait_tensorcnt(0);
    }
#else
    for (int j = tid; j < KTOT; j += 256) {
        ksr[j] = (j >= KPAD && j < LEN + KPAD) ? kbr[(size_t)c * LEN + (j - KPAD)]
                                               : (__bf16)0.0f;
    }
#endif
    __syncthreads();

    const int wave = tid >> 5;
    const int lane = tid & 31;
    const int iG   = (blockIdx.y * 8 + wave) * 8;   // first of 8 output tiles (even)
    const int m    = lane & 15;                     // A row / D column
    const int g    = lane >> 4;                     // lane-group

    v8f acc0 = {}, acc1 = {}, acc2 = {}, acc3 = {};
    v8f acc4 = {}, acc5 = {}, acc6 = {}, acc7 = {};

    const int bsel = (m < BATCH) ? m : 0;           // cols 8..15 are discarded
    const __bf16* xrow = xb + ((size_t)bsel * C_CH + c) * LEN + g * 16;

#define LOADB(JP)  (*(const v16bf*)(xrow + (size_t)(JP) * 32))

    // A[e] for element e reads ksr at ascending offsets (no byte reversal):
    //   d'(e) = e + 16g;  A[e] = ksr[rbase + e]
#define BUILD_A(A, DELTA)                                                      \
    {                                                                          \
        int rbase = (LEN + KPAD - 1) - 16 * (DELTA) - m + 16 * g;              \
        _Pragma("unroll")                                                      \
        for (int e = 0; e < 16; ++e)                                           \
            A[e] = ksr[rbase + e];                                             \
    }

#define WMMA(ACC, A, B)                                                        \
    ACC = __builtin_amdgcn_wmma_f32_16x16x32_bf16(false, A, false, B,          \
                                                  (short)0, ACC, false, false);

    // ---- even deltas, main (delta <= iG): all 4 even tiles active ----------
    {
        int jp0 = iG >> 1;
        v16bf B0 = LOADB(jp0 + 0), B1 = LOADB(jp0 + 1);
        v16bf B2 = LOADB(jp0 + 2), B3 = LOADB(jp0 + 3);
        for (int delta = 0;; delta += 2) {
            v16bf A; BUILD_A(A, delta)
            WMMA(acc0, A, B0) WMMA(acc2, A, B1)
            WMMA(acc4, A, B2) WMMA(acc6, A, B3)
            if (delta >= iG) break;
            B3 = B2; B2 = B1; B1 = B0;
            --jp0;
            B0 = LOADB(jp0);                 // consumed next iteration
        }
    }
    // ---- odd deltas, main (delta <= iG+1): all 4 odd tiles active ----------
    {
        int jp0 = iG >> 1;
        v16bf B0 = LOADB(jp0 + 0), B1 = LOADB(jp0 + 1);
        v16bf B2 = LOADB(jp0 + 2), B3 = LOADB(jp0 + 3);
        for (int delta = 1;; delta += 2) {
            v16bf A; BUILD_A(A, delta)
            WMMA(acc1, A, B0) WMMA(acc3, A, B1)
            WMMA(acc5, A, B2) WMMA(acc7, A, B3)
            if (delta >= iG + 1) break;
            B3 = B2; B2 = B1; B1 = B0;
            --jp0;
            B0 = LOADB(jp0);
        }
    }

    // ---- guarded tails: delta in (iG+1 .. iG+7], few iterations ------------
#define TAIL_WMMA(ACC, T, DELTA)                                               \
    {                                                                          \
        if (iG + (T) >= (DELTA)) {                                             \
            int jp = (iG + (T) - (DELTA)) >> 1;                                \
            v16bf B = LOADB(jp);                                               \
            WMMA(ACC, A, B)                                                    \
        }                                                                      \
    }
    for (int delta = iG + 2; delta <= iG + 6; delta += 2) {
        v16bf A; BUILD_A(A, delta)
        TAIL_WMMA(acc2, 2, delta) TAIL_WMMA(acc4, 4, delta) TAIL_WMMA(acc6, 6, delta)
    }
    for (int delta = iG + 3; delta <= iG + 7; delta += 2) {
        v16bf A; BUILD_A(A, delta)
        TAIL_WMMA(acc3, 3, delta) TAIL_WMMA(acc5, 5, delta) TAIL_WMMA(acc7, 7, delta)
    }
#undef TAIL_WMMA
#undef WMMA
#undef BUILD_A
#undef LOADB

    // ---- store: D element r of lane l is Y[row 8*g + r, col l&15] ----------
    if (m < BATCH) {
        float* obase = out + ((size_t)m * C_CH + c) * LEN + g * 8;
#define STORE_ACC(ACC, T)                                                      \
        {                                                                      \
            float* op = obase + (size_t)(iG + (T)) * 16;                       \
            *(float4*)(op + 0) = make_float4(ACC[0], ACC[1], ACC[2], ACC[3]);  \
            *(float4*)(op + 4) = make_float4(ACC[4], ACC[5], ACC[6], ACC[7]);  \
        }
        STORE_ACC(acc0, 0) STORE_ACC(acc1, 1) STORE_ACC(acc2, 2) STORE_ACC(acc3, 3)
        STORE_ACC(acc4, 4) STORE_ACC(acc5, 5) STORE_ACC(acc6, 6) STORE_ACC(acc7, 7)
#undef STORE_ACC
    }
}

extern "C" void kernel_launch(void* const* d_in, const int* in_sizes, int n_in,
                              void* d_out, int out_size, void* d_ws, size_t ws_size,
                              hipStream_t stream) {
    const float* x    = (const float*)d_in[0];   // (8, 256, 8192) f32
    const float* kern = (const float*)d_in[1];   // (256, 8192)    f32
    float* out = (float*)d_out;                  // (8, 256, 8192) f32

    // workspace: [0,4MB) reversed khat bf16 ; [4MB,36MB) x bf16
    __bf16* kbr = (__bf16*)d_ws;
    __bf16* xb  = (__bf16*)((char*)d_ws + (size_t)C_CH * LEN * sizeof(__bf16));

    prep_kern<<<dim3((C_CH * LEN) / 256), 256, 0, stream>>>(kern, kbr);

    int nx = BATCH * C_CH * LEN;
    cvt_x<<<dim3(nx / (256 * 4)), 256, 0, stream>>>(x, xb, nx);

    conv_kern<<<dim3(C_CH, 8), 256, 0, stream>>>(xb, kbr, out);
}
